// AgentEncoder_21337397527174
// MI455X (gfx1250) — matrix operand order repaired and checked by reading, not compiled, not run on hardware
//
#include <hip/hip_runtime.h>
#include <hip/hip_bf16.h>
#include <math.h>

// ---------------------------------------------------------------------------
// AgentEncoder for MI455X (gfx1250, wave32, WMMA).
// All dense algebra runs through v_wmma_f32_16x16x32_f16 (f16 in, f32 acc).
// rel_emb mean is commuted through the linear layer: mean(silu(...)) @ re_w2.
// GEMM: 128x64 block tile, 8 waves, each wave a 32x32 register tile ->
// 4 WMMAs per 4 fragment loads per K-step (matrix-pipe bound, not LDS bound).
// Input order assumption: setup_inputs() dict insertion order, nested dicts
// expanded in place (54 inputs total; see index table in kernel_launch).
// ---------------------------------------------------------------------------

typedef __attribute__((ext_vector_type(16))) _Float16 v16h;
typedef __attribute__((ext_vector_type(8)))  _Float16 v8h;
typedef __attribute__((ext_vector_type(8)))  float    v8f;

#define Bb 4
#define Nn 128
#define Tt 256
#define Dd 768
#define Hh 12
#define DH 64
#define FF 3072
#define ROWS (Bb * Nn)      // 512
#define MAPROWS (Bb * Tt)   // 1024

__device__ __forceinline__ float dev_silu(float x) { return x / (1.0f + __expf(-x)); }
__device__ __forceinline__ float dev_gelu(float x) { return 0.5f * x * (1.0f + erff(x * 0.70710678118654752f)); }

template <int ACT>
__device__ __forceinline__ float apply_act(float x) {
    if (ACT == 1) return dev_silu(x);
    if (ACT == 2) return dev_gelu(x);
    return x;
}

// concat two aligned 8-half LDS chunks into one fragment (register renaming only)
__device__ __forceinline__ v16h frag_cat(const _Float16* lo, const _Float16* hi) {
    v8h a = *(const v8h*)lo;
    v8h b = *(const v8h*)hi;
    return __builtin_shufflevector(a, b, 0, 1, 2, 3, 4, 5, 6, 7, 8, 9, 10, 11, 12, 13, 14, 15);
}

// ---------------------------------------------------------------------------
// Tiled WMMA GEMM:  out[M,N] = act(A[M,K] @ W[K,N] + bias[N]) (+ res)
// Block: 256 threads (8 waves), 128x64 block tile, BK = 32.
// Wave w: m-slot (w>>1)*32, n-slot (w&1)*32 -> 32x32 register tile,
// 4 accumulators, 4 WMMAs per K-step. Requires M%128==0, N%64==0, K%32==0.
// Fragment layouts follow CDNA5 ISA 7.12.2 (16-bit A 16x32, B 32x16, f32 C).
// LDS rows padded to 40 halfs (80 B) so all fragment chunks stay 16B-aligned.
// ---------------------------------------------------------------------------
template <int ACT, bool RES>
__global__ __launch_bounds__(256) void gemm_kernel(
    const float* __restrict__ A, const float* __restrict__ W,
    const float* __restrict__ bias, const float* __restrict__ res,
    float* __restrict__ out, int M, int N, int K)
{
    __shared__ _Float16 Ah[128][40];  // [m][k]
    __shared__ _Float16 Bt[64][40];   // [n][k] (W tile transposed)

    const int tid  = threadIdx.x;
    const int lane = tid & 31;
    const int wave = tid >> 5;
    const int l15  = lane & 15;
    const int lhi  = lane >> 4;

    const int bm = blockIdx.x * 128;
    const int bn = blockIdx.y * 64;
    const int wm = (wave >> 1) * 32;  // 0,32,64,96
    const int wn = (wave & 1) * 32;   // 0,32

    const int arow = tid >> 1;          // 0..127
    const int acol = (tid & 1) * 16;    // 0,16
    const int bkk  = tid & 31;          // 0..31
    const int bng  = (tid >> 5) * 8;    // 0..56

    v8f acc00 = {}, acc01 = {}, acc10 = {}, acc11 = {};

    for (int k0 = 0; k0 < K; k0 += 32) {
        // ---- stage A (128x32) and W (32x64 transposed) into LDS as f16 ----
        {
            const float4* ap = (const float4*)(A + (size_t)(bm + arow) * K + (k0 + acol));
            const float4* bp = (const float4*)(W + (size_t)(k0 + bkk) * N + (bn + bng));
            if (k0 + 32 < K) {
                __builtin_prefetch((const float*)ap + 32, 0, 1);       // global_prefetch
                __builtin_prefetch((const float*)bp + (size_t)32 * N, 0, 1);
            }
            alignas(16) _Float16 tmp[16];
#pragma unroll
            for (int j = 0; j < 4; ++j) {
                const float4 f = ap[j];
                tmp[4 * j + 0] = (_Float16)f.x;
                tmp[4 * j + 1] = (_Float16)f.y;
                tmp[4 * j + 2] = (_Float16)f.z;
                tmp[4 * j + 3] = (_Float16)f.w;
            }
            *(v8h*)&Ah[arow][acol]     = *(const v8h*)&tmp[0];
            *(v8h*)&Ah[arow][acol + 8] = *(const v8h*)&tmp[8];

            const float4 g0 = bp[0], g1 = bp[1];
            Bt[bng + 0][bkk] = (_Float16)g0.x;
            Bt[bng + 1][bkk] = (_Float16)g0.y;
            Bt[bng + 2][bkk] = (_Float16)g0.z;
            Bt[bng + 3][bkk] = (_Float16)g0.w;
            Bt[bng + 4][bkk] = (_Float16)g1.x;
            Bt[bng + 5][bkk] = (_Float16)g1.y;
            Bt[bng + 6][bkk] = (_Float16)g1.z;
            Bt[bng + 7][bkk] = (_Float16)g1.w;
        }
        __syncthreads();

        // ---- fragments + 4 WMMAs ----
        const int kg = lhi * 8;   // A: lanes 0-15 hold k 0-7/16-23, 16-31 hold 8-15/24-31
        const int kb = lhi * 8;   // B: same k order as A (0-7/16-23, 8-15/24-31)
        const v16h a0 = frag_cat(&Ah[wm + l15][kg],      &Ah[wm + l15][kg + 16]);
        const v16h a1 = frag_cat(&Ah[wm + 16 + l15][kg], &Ah[wm + 16 + l15][kg + 16]);
        const v16h b0 = frag_cat(&Bt[wn + l15][kb],      &Bt[wn + l15][kb + 16]);
        const v16h b1 = frag_cat(&Bt[wn + 16 + l15][kb], &Bt[wn + 16 + l15][kb + 16]);

        acc00 = __builtin_amdgcn_wmma_f32_16x16x32_f16(false, a0, false, b0, (short)0, acc00, false, false);
        acc01 = __builtin_amdgcn_wmma_f32_16x16x32_f16(false, a0, false, b1, (short)0, acc01, false, false);
        acc10 = __builtin_amdgcn_wmma_f32_16x16x32_f16(false, a1, false, b0, (short)0, acc10, false, false);
        acc11 = __builtin_amdgcn_wmma_f32_16x16x32_f16(false, a1, false, b1, (short)0, acc11, false, false);
        __syncthreads();
    }

    // ---- epilogue: lane holds n = l15, m = lhi*8 + j per 16x16 C tile ----
    const int n0 = bn + wn + l15;
    const int n1 = n0 + 16;
    const float bv0 = bias ? bias[n0] : 0.0f;
    const float bv1 = bias ? bias[n1] : 0.0f;
#pragma unroll
    for (int j = 0; j < 8; ++j) {
        const int m0 = bm + wm + lhi * 8 + j;
        const int m1 = m0 + 16;

        float v = apply_act<ACT>(acc00[j] + bv0);
        if (RES) v += res[(size_t)m0 * N + n0];
        out[(size_t)m0 * N + n0] = v;

        v = apply_act<ACT>(acc01[j] + bv1);
        if (RES) v += res[(size_t)m0 * N + n1];
        out[(size_t)m0 * N + n1] = v;

        v = apply_act<ACT>(acc10[j] + bv0);
        if (RES) v += res[(size_t)m1 * N + n0];
        out[(size_t)m1 * N + n0] = v;

        v = apply_act<ACT>(acc11[j] + bv1);
        if (RES) v += res[(size_t)m1 * N + n1];
        out[(size_t)m1 * N + n1] = v;
    }
}

// ---------------------------------------------------------------------------
// Multi-head attention core. Q:[B*Lq,768] K,V:[B*Lk,768] laid out (L, H*DH).
// Block: 64 threads (2 waves); each wave owns one 16-query tile.
// Grid: B * H * (Lq/32).  Scores staged in LDS f32, softmax via shfl (wave32).
// Both QK^T and P@V are WMMA f16->f32.
// ---------------------------------------------------------------------------
__global__ __launch_bounds__(64) void attn_kernel(
    const float* __restrict__ Qm, const float* __restrict__ Km,
    const float* __restrict__ Vm, float* __restrict__ Om,
    int Lq, int Lk, float scale)
{
    __shared__ float Sw[2][16][256];

    const int wave = threadIdx.x >> 5;
    const int lane = threadIdx.x & 31;
    const int l15  = lane & 15;
    const int lhi  = lane >> 4;
    const int kg   = lhi * 8;

    const int qtiles2 = Lq / 32;
    const int qt2 = blockIdx.x % qtiles2;
    const int bh  = blockIdx.x / qtiles2;
    const int b   = bh / Hh;
    const int h   = bh % Hh;
    const int q0  = qt2 * 32 + wave * 16;

    const float* Qp = Qm + (size_t)b * Lq * Dd + h * DH;
    const float* Kp = Km + (size_t)b * Lk * Dd + h * DH;
    const float* Vp = Vm + (size_t)b * Lk * Dd + h * DH;
    float*       Op = Om + (size_t)b * Lq * Dd + h * DH;

    // Q fragments (16 queries x 64 dh, two K=32 steps)
    v16h qa[2];
#pragma unroll
    for (int s = 0; s < 2; ++s)
#pragma unroll
        for (int j = 0; j < 16; ++j) {
            const int dh = s * 32 + ((j >> 3) * 16) + kg + (j & 7);
            qa[s][j] = (_Float16)Qp[(size_t)(q0 + l15) * Dd + dh];
        }

    float (*S)[256] = Sw[wave];

    // S = scale * Q @ K^T
    for (int kt = 0; kt < Lk / 16; ++kt) {
        v8f acc = {};
        const int key = kt * 16 + l15;        // B-fragment column index
#pragma unroll
        for (int s = 0; s < 2; ++s) {
            v16h kf;
#pragma unroll
            for (int j = 0; j < 16; ++j) {
                const int dh = s * 32 + ((j >> 3) * 16) + kg + (j & 7);
                kf[j] = (_Float16)Kp[(size_t)key * Dd + dh];
            }
            acc = __builtin_amdgcn_wmma_f32_16x16x32_f16(false, qa[s], false, kf, (short)0, acc, false, false);
        }
#pragma unroll
        for (int j = 0; j < 8; ++j)
            S[lhi * 8 + j][kt * 16 + l15] = acc[j] * scale;
    }
    __syncthreads();

    // row-wise softmax over Lk (each wave owns its own 16 rows)
    for (int r = 0; r < 16; ++r) {
        float m = -1e30f;
        for (int c = lane; c < Lk; c += 32) m = fmaxf(m, S[r][c]);
#pragma unroll
        for (int off = 16; off; off >>= 1) m = fmaxf(m, __shfl_xor(m, off));
        float sum = 0.0f;
        for (int c = lane; c < Lk; c += 32) {
            const float e = __expf(S[r][c] - m);
            S[r][c] = e;
            sum += e;
        }
#pragma unroll
        for (int off = 16; off; off >>= 1) sum += __shfl_xor(sum, off);
        const float inv = 1.0f / sum;
        for (int c = lane; c < Lk; c += 32) S[r][c] *= inv;
    }
    __syncthreads();

    // O = P @ V  (contraction over keys)
    for (int nt = 0; nt < 4; ++nt) {
        v8f acc = {};
        const int dh = nt * 16 + l15;         // B-fragment column = dh
        for (int ks = 0; ks < Lk / 32; ++ks) {
            v16h af, bfv;
#pragma unroll
            for (int j = 0; j < 16; ++j) {
                const int key = ks * 32 + ((j >> 3) * 16) + kg + (j & 7);
                af[j] = (_Float16)S[l15][key];
            }
#pragma unroll
            for (int j = 0; j < 16; ++j) {
                const int key = ks * 32 + ((j >> 3) * 16) + kg + (j & 7);
                bfv[j] = (_Float16)Vp[(size_t)key * Dd + dh];
            }
            acc = __builtin_amdgcn_wmma_f32_16x16x32_f16(false, af, false, bfv, (short)0, acc, false, false);
        }
#pragma unroll
        for (int j = 0; j < 8; ++j)
            Op[(size_t)(q0 + lhi * 8 + j) * Dd + nt * 16 + l15] = acc[j];
    }
}

// ---------------------------------------------------------------------------
// h = silu(agent_states @ se_w1 + se_b1)   (K=5 tiny GEMM, VALU)
// ---------------------------------------------------------------------------
__global__ __launch_bounds__(256) void se_kernel(
    const float* __restrict__ agent, const float* __restrict__ w1,
    const float* __restrict__ b1, float* __restrict__ out)
{
    const int row = blockIdx.x;       // 0..511
    __shared__ float a[5];
    if (threadIdx.x < 5) a[threadIdx.x] = agent[row * 5 + threadIdx.x];
    __syncthreads();
    for (int d = threadIdx.x; d < Dd; d += 256) {
        float v = b1[d];
#pragma unroll
        for (int f = 0; f < 5; ++f) v += a[f] * w1[f * Dd + d];
        out[(size_t)row * Dd + d] = dev_silu(v);
    }
}

// ---------------------------------------------------------------------------
// relmean[b,i,:] = mean_k silu(relfeat(b,i,k) @ re_w1 + re_b1)
// ---------------------------------------------------------------------------
__global__ __launch_bounds__(256) void rel_kernel(
    const float* __restrict__ agent, const float* __restrict__ w1,
    const float* __restrict__ b1, float* __restrict__ out)
{
    const int bi = blockIdx.x;        // b*128 + i
    const int b  = bi >> 7;
    const int i  = bi & 127;
    __shared__ float ag[Nn][5];
    for (int idx = threadIdx.x; idx < Nn * 5; idx += 256)
        ag[idx / 5][idx % 5] = agent[((size_t)b * Nn + idx / 5) * 5 + idx % 5];
    __syncthreads();

    const float px = ag[i][0], py = ag[i][1];
    const float vx = ag[i][2], vy = ag[i][3];
    const float hd = ag[i][4];

    const int d0 = threadIdx.x, d1 = threadIdx.x + 256, d2 = threadIdx.x + 512;
    float acc0 = 0.0f, acc1 = 0.0f, acc2 = 0.0f;

    for (int k = 0; k < Nn; ++k) {
        const float dpx = ag[k][0] - px, dpy = ag[k][1] - py;
        const float dvx = ag[k][2] - vx, dvy = ag[k][3] - vy;
        const float dist = sqrtf(dpx * dpx + dpy * dpy);
        const float rsp  = sqrtf(dvx * dvx + dvy * dvy);
        const float dot  = dpx * dvx + dpy * dvy;
        const float ttc  = (dot < 0.0f) ? fminf(fmaxf(dist / (rsp + 1e-3f), 0.0f), 10.0f) : 10.0f;
        const float brg  = atan2f(dpy, dpx) - hd;
        const float f[6] = {dpx, dpy, dvx, dvy, ttc, brg};
        float s0 = b1[d0], s1 = b1[d1], s2 = b1[d2];
#pragma unroll
        for (int t = 0; t < 6; ++t) {
            s0 += f[t] * w1[t * Dd + d0];
            s1 += f[t] * w1[t * Dd + d1];
            s2 += f[t] * w1[t * Dd + d2];
        }
        acc0 += dev_silu(s0);
        acc1 += dev_silu(s1);
        acc2 += dev_silu(s2);
    }
    const float inv = 1.0f / (float)Nn;
    out[(size_t)bi * Dd + d0] = acc0 * inv;
    out[(size_t)bi * Dd + d1] = acc1 * inv;
    out[(size_t)bi * Dd + d2] = acc2 * inv;
}

// ---------------------------------------------------------------------------
// LayerNorm over last dim (768), one block per row.
// ---------------------------------------------------------------------------
__global__ __launch_bounds__(256) void ln_kernel(
    const float* __restrict__ in, const float* __restrict__ g,
    const float* __restrict__ bt, float* __restrict__ out)
{
    const int row = blockIdx.x;
    const float* x = in + (size_t)row * Dd;
    const int tid = threadIdx.x, lane = tid & 31, wv = tid >> 5;
    __shared__ float red[8];

    const float v0 = x[tid], v1 = x[tid + 256], v2 = x[tid + 512];

    float s = v0 + v1 + v2;
#pragma unroll
    for (int off = 16; off; off >>= 1) s += __shfl_xor(s, off);
    if (lane == 0) red[wv] = s;
    __syncthreads();
    float tot = 0.0f;
#pragma unroll
    for (int i = 0; i < 8; ++i) tot += red[i];
    const float mean = tot * (1.0f / (float)Dd);
    __syncthreads();

    const float dA = v0 - mean, dB = v1 - mean, dC = v2 - mean;
    float sv = dA * dA + dB * dB + dC * dC;
#pragma unroll
    for (int off = 16; off; off >>= 1) sv += __shfl_xor(sv, off);
    if (lane == 0) red[wv] = sv;
    __syncthreads();
    float vt = 0.0f;
#pragma unroll
    for (int i = 0; i < 8; ++i) vt += red[i];
    const float inv = rsqrtf(vt * (1.0f / (float)Dd) + 1e-5f);

    out[(size_t)row * Dd + tid]       = dA * inv * g[tid]       + bt[tid];
    out[(size_t)row * Dd + tid + 256] = dB * inv * g[tid + 256] + bt[tid + 256];
    out[(size_t)row * Dd + tid + 512] = dC * inv * g[tid + 512] + bt[tid + 512];
}

// ---------------------------------------------------------------------------
// logits = x @ cls_w + cls_b, broadcast to (B,N,N,6).
// ---------------------------------------------------------------------------
__global__ __launch_bounds__(192) void cls_kernel(
    const float* __restrict__ x, const float* __restrict__ w,
    const float* __restrict__ bz, float* __restrict__ outI)
{
    const int bi = blockIdx.x;        // b*128 + i
    __shared__ float lg[6];
    const int g = threadIdx.x >> 5;   // 0..5
    const int lane = threadIdx.x & 31;
    const float* xr = x + (size_t)bi * Dd;
    float acc = 0.0f;
    for (int d = lane; d < Dd; d += 32) acc += xr[d] * w[d * 6 + g];
#pragma unroll
    for (int off = 16; off; off >>= 1) acc += __shfl_xor(acc, off);
    if (lane == 0) lg[g] = acc + bz[g];
    __syncthreads();
    float* dst = outI + (size_t)bi * Nn * 6;
    for (int idx = threadIdx.x; idx < Nn * 6; idx += 192) dst[idx] = lg[idx % 6];
}

// ---------------------------------------------------------------------------
// host-side orchestration
// ---------------------------------------------------------------------------
extern "C" void kernel_launch(void* const* d_in, const int* in_sizes, int n_in,
                              void* d_out, int out_size, void* d_ws, size_t ws_size,
                              hipStream_t stream)
{
    (void)in_sizes; (void)n_in; (void)out_size; (void)ws_size;

    const float* agent = (const float*)d_in[0];
    const float* mapf  = (const float*)d_in[1];
#define PW(i) ((const float*)d_in[(i)])
    const float *se_w1 = PW(2),  *se_b1 = PW(3),  *se_w2 = PW(4),  *se_b2 = PW(5);
    const float *sp_wq = PW(6),  *sp_bq = PW(7),  *sp_wk = PW(8),  *sp_bk = PW(9);
    const float *sp_wv = PW(10), *sp_bv = PW(11), *sp_wo = PW(12), *sp_bo = PW(13);
    const float *sp_ln_g = PW(14), *sp_ln_b = PW(15);
    const float *re_w1 = PW(16), *re_b1 = PW(17), *re_w2 = PW(18), *re_b2 = PW(19);
    const float *ra_wq = PW(20), *ra_bq = PW(21), *ra_wk = PW(22), *ra_bk = PW(23);
    const float *ra_wv = PW(24), *ra_bv = PW(25), *ra_wo = PW(26), *ra_bo = PW(27);
    const float *ra_ln_g = PW(28), *ra_ln_b = PW(29);
    const float *f2_w1 = PW(30), *f2_b1 = PW(31), *f2_w2 = PW(32), *f2_b2 = PW(33);
    const float *f2_ln_g = PW(34), *f2_ln_b = PW(35);
    const float *a_wq = PW(36), *a_bq = PW(37), *a_wk = PW(38), *a_bk = PW(39);
    const float *a_wv = PW(40), *a_bv = PW(41), *a_wo = PW(42), *a_bo = PW(43);
    const float *a2l_ln_g = PW(44), *a2l_ln_b = PW(45);
    const float *f3_w1 = PW(46), *f3_b1 = PW(47), *f3_w2 = PW(48), *f3_b2 = PW(49);
    const float *f3_ln_g = PW(50), *f3_ln_b = PW(51);
    const float *cls_w = PW(52), *cls_b = PW(53);
#undef PW

    const size_t XE = (size_t)ROWS * Dd;           // 393216
    float* X  = (float*)d_ws;                      // current activations
    float* Y  = X  + XE;                           // staging / qin / level2
    float* Qb = Y  + XE;                           // Q projections
    float* AO = Qb + XE;                           // attention output
    float* N2 = AO + XE;                           // LN temp / rel mean
    float* Kb = N2 + XE;                           // K projections (up to 1024 rows)
    float* Vb = Kb + (size_t)MAPROWS * Dd;         // V projections
    float* Hf = Vb + (size_t)MAPROWS * Dd;         // FF hidden (512x3072)

    float* out_l1 = (float*)d_out;
    float* out_l2 = out_l1 + XE;
    float* out_l3 = out_l2 + XE;
    float* out_it = out_l3 + XE;

    const dim3 g768(ROWS / 128, Dd / 64);          // (4,12)  512x768 GEMM
    const dim3 gFF1(ROWS / 128, FF / 64);          // (4,48)  512x3072
    const dim3 gMap(MAPROWS / 128, Dd / 64);       // (8,12)  1024x768
    const dim3 attG(Bb * Hh * (Nn / 32));          // 192 blocks
    const float scale = 0.125f;                    // 1/sqrt(64)

    // ---- state embed + level1 ----
    se_kernel<<<ROWS, 256, 0, stream>>>(agent, se_w1, se_b1, Hf);
    gemm_kernel<0, false><<<g768, 256, 0, stream>>>(Hf, se_w2, se_b2, nullptr, X, ROWS, Dd, Dd);
    hipMemcpyAsync(out_l1, X, XE * sizeof(float), hipMemcpyDeviceToDevice, stream);

    // ---- spatial MHA + LN ----
    gemm_kernel<0, false><<<g768, 256, 0, stream>>>(X, sp_wq, sp_bq, nullptr, Qb, ROWS, Dd, Dd);
    gemm_kernel<0, false><<<g768, 256, 0, stream>>>(X, sp_wk, sp_bk, nullptr, Kb, ROWS, Dd, Dd);
    gemm_kernel<0, false><<<g768, 256, 0, stream>>>(X, sp_wv, sp_bv, nullptr, Vb, ROWS, Dd, Dd);
    attn_kernel<<<attG, 64, 0, stream>>>(Qb, Kb, Vb, AO, Nn, Nn, scale);
    gemm_kernel<0, true><<<g768, 256, 0, stream>>>(AO, sp_wo, sp_bo, X, Y, ROWS, Dd, Dd);
    ln_kernel<<<ROWS, 256, 0, stream>>>(Y, sp_ln_g, sp_ln_b, X);

    // ---- relational context (mean commuted through linear layer) ----
    rel_kernel<<<ROWS, 256, 0, stream>>>(agent, re_w1, re_b1, N2);
    gemm_kernel<0, true><<<g768, 256, 0, stream>>>(N2, re_w2, re_b2, X, Y, ROWS, Dd, Dd); // Y = x + rel_context

    // ---- relational MHA + LN ----
    gemm_kernel<0, false><<<g768, 256, 0, stream>>>(Y, ra_wq, ra_bq, nullptr, Qb, ROWS, Dd, Dd);
    gemm_kernel<0, false><<<g768, 256, 0, stream>>>(X, ra_wk, ra_bk, nullptr, Kb, ROWS, Dd, Dd);
    gemm_kernel<0, false><<<g768, 256, 0, stream>>>(X, ra_wv, ra_bv, nullptr, Vb, ROWS, Dd, Dd);
    attn_kernel<<<attG, 64, 0, stream>>>(Qb, Kb, Vb, AO, Nn, Nn, scale);
    gemm_kernel<0, true><<<g768, 256, 0, stream>>>(AO, ra_wo, ra_bo, X, Y, ROWS, Dd, Dd);
    ln_kernel<<<ROWS, 256, 0, stream>>>(Y, ra_ln_g, ra_ln_b, X);

    // ---- FF block 2 (pre-LN, gelu) -> level2 ----
    ln_kernel<<<ROWS, 256, 0, stream>>>(X, f2_ln_g, f2_ln_b, N2);
    gemm_kernel<2, false><<<gFF1, 256, 0, stream>>>(N2, f2_w1, f2_b1, nullptr, Hf, ROWS, FF, Dd);
    gemm_kernel<0, true><<<g768, 256, 0, stream>>>(Hf, f2_w2, f2_b2, X, Y, ROWS, Dd, FF); // Y = level2
    hipMemcpyAsync(out_l2, Y, XE * sizeof(float), hipMemcpyDeviceToDevice, stream);

    // ---- interaction logits broadcast ----
    cls_kernel<<<ROWS, 192, 0, stream>>>(Y, cls_w, cls_b, out_it);

    // ---- agent-to-lane cross attention + LN ----
    gemm_kernel<0, false><<<g768, 256, 0, stream>>>(Y, a_wq, a_bq, nullptr, Qb, ROWS, Dd, Dd);
    gemm_kernel<0, false><<<gMap, 256, 0, stream>>>(mapf, a_wk, a_bk, nullptr, Kb, MAPROWS, Dd, Dd);
    gemm_kernel<0, false><<<gMap, 256, 0, stream>>>(mapf, a_wv, a_bv, nullptr, Vb, MAPROWS, Dd, Dd);
    attn_kernel<<<attG, 64, 0, stream>>>(Qb, Kb, Vb, AO, Nn, Tt, scale);
    gemm_kernel<0, true><<<g768, 256, 0, stream>>>(AO, a_wo, a_bo, Y, X, ROWS, Dd, Dd);
    ln_kernel<<<ROWS, 256, 0, stream>>>(X, a2l_ln_g, a2l_ln_b, Y);

    // ---- FF block 3 -> level3 ----
    ln_kernel<<<ROWS, 256, 0, stream>>>(Y, f3_ln_g, f3_ln_b, N2);
    gemm_kernel<2, false><<<gFF1, 256, 0, stream>>>(N2, f3_w1, f3_b1, nullptr, Hf, ROWS, FF, Dd);
    gemm_kernel<0, true><<<g768, 256, 0, stream>>>(Hf, f3_w2, f3_b2, Y, X, ROWS, Dd, FF); // X = level3
    hipMemcpyAsync(out_l3, X, XE * sizeof(float), hipMemcpyDeviceToDevice, stream);
}
